// RegionProposalNetwork1d_43430709297800
// MI455X (gfx1250) — matrix operand order repaired and checked by reading, not compiled, not run on hardware
//
#include <hip/hip_runtime.h>
#include <math.h>

#define LSEQ   262144
#define NANCH  (5*LSEQ)
#define PRE_N  6000
#define POST_N 300
#define SORT_N 8192

typedef __attribute__((ext_vector_type(16))) __bf16 v16bf;
typedef __attribute__((ext_vector_type(8)))  float  v8f;

// ---------------------------------------------------------------- helpers
__device__ __forceinline__ __bf16 f2bf(float f){
  unsigned u = __float_as_uint(f);
  unsigned r = (u + 0x7FFFu + ((u >> 16) & 1u)) >> 16;
  unsigned short h = (unsigned short)r;
  return __builtin_bit_cast(__bf16, h);
}

__device__ __forceinline__ void atomicMaxF(float* addr, float v){
  if (v >= 0.0f) atomicMax((int*)addr, __float_as_int(v));
  else           atomicMin((unsigned int*)addr, __float_as_uint(v));
}

__device__ __forceinline__ float iou1d(float as, float ae, float aw,
                                       float gs, float ge, float gw){
  float inter = fminf(ae, ge) - fmaxf(as, gs) + 1.0f;
  inter = fmaxf(inter, 0.0f);
  return inter / (aw + gw - inter);
}

// ---------------------------------------------------------------- generic fill
__global__ void fill_kernel(float* p, float v, int n){
  for (int i = blockIdx.x*blockDim.x + threadIdx.x; i < n; i += gridDim.x*blockDim.x) p[i] = v;
}

// ---------------------------------------------------------------- depthwise 3-tap dilated conv
__global__ void dwconv_kernel(const float* __restrict__ X, const float* __restrict__ W,
                              float* __restrict__ Y, int pad, int dil, int n){
  for (int idx = blockIdx.x*blockDim.x + threadIdx.x; idx < n; idx += gridDim.x*blockDim.x){
    int c = idx >> 18;            // / LSEQ
    int l = idx & (LSEQ - 1);
    const float* xr = X + (size_t)c*LSEQ;
    const float* wr = W + c*3;
    if (l + 8192 < LSEQ) __builtin_prefetch(xr + l + 8192, 0, 1);  // global_prefetch_b8
    float acc = 0.0f;
    #pragma unroll
    for (int k = 0; k < 3; ++k){
      int p = l - pad + k*dil;
      if (p >= 0 && p < LSEQ) acc += wr[k]*xr[p];
    }
    Y[idx] = acc;
  }
}

// ---------------------------------------------------------------- pointwise (1x1) conv as WMMA GEMM
// Y[COUT, L] = W[COUT, CIN] * X[CIN, L] + bias   (ACT: 0=none 1=relu 2=sigmoid)
// Per wave: one 16-row M-tile x four 16-col N-tiles. X is staged into a private
// 32x64 f32 LDS tile per K-step via GLOBAL_LOAD_ASYNC_TO_LDS_B128 (coalesced b128
// along L, ASYNCcnt-tracked), then bf16 B-fragments are built from LDS and fed to
// V_WMMA_F32_16X16X32_BF16 with f32 accumulation.
template<int CIN, int COUT, int ACT>
__global__ void pwconv_wmma_t(const float* __restrict__ X, const float* __restrict__ W,
                              const float* __restrict__ bias, float* __restrict__ Y){
  constexpr int KSTEPS = (CIN + 31)/32;
  constexpr int MTILES = (COUT + 15)/16;
  constexpr int NT = 4;                       // 64 columns per wave
  __shared__ __align__(16) float tile[8][32*64];

  const int lane = threadIdx.x & 31;
  const int wv   = threadIdx.x >> 5;
  const int wid  = blockIdx.x*8 + wv;
  const int ngroups = LSEQ/(16*NT);
  if (wid >= MTILES*ngroups) return;
  const int mt   = wid / ngroups;
  const int ng   = wid % ngroups;
  const int m0   = mt*16;
  const int col0 = ng*(16*NT);

  // ---- A fragments (weights, 16x32 bf16 per ISA 7.12.2), fully unrolled
  v16bf afrag[KSTEPS];
  {
    const int am    = m0 + (lane & 15);
    const int khalf = (lane >> 4) << 3;       // 0 or 8
    #pragma unroll
    for (int ks = 0; ks < KSTEPS; ++ks){
      v16bf a;
      #pragma unroll
      for (int j = 0; j < 8; ++j){
        int kb = (j < 4) ? (2*j + khalf) : (16 + 2*(j - 4) + khalf);
        int k0 = ks*32 + kb;
        float w0 = 0.0f, w1 = 0.0f;
        if (am < COUT){
          if (k0     < CIN) w0 = W[am*CIN + k0];
          if (k0 + 1 < CIN) w1 = W[am*CIN + k0 + 1];
        }
        a[2*j]     = f2bf(w0);
        a[2*j + 1] = f2bf(w1);
      }
      afrag[ks] = a;
    }
  }

  const int bn    = lane & 15;
  const int bko   = (lane >> 4) << 3;         // 0 or 8 (K offset for B, table order)
  const int mbase = m0 + ((lane >> 4) << 3);  // D layout: lanes>=16 hold M=8..15

  float* tw = &tile[wv][0];
  // flat shared pointer: low 32 bits == LDS byte offset (aperture truncation)
  const unsigned ldsbase = (unsigned)(size_t)tw;

  v8f acc[NT];
  #pragma unroll
  for (int t = 0; t < NT; ++t) acc[t] = (v8f){0.f,0.f,0.f,0.f,0.f,0.f,0.f,0.f};

  #pragma unroll
  for (int ks = 0; ks < KSTEPS; ++ks){
    // ---- async-stage 32 rows x 64 cols f32 (rows clamped; masked at use)
    #pragma unroll
    for (int p = 0; p < 16; ++p){
      int q  = lane + 32*p;                   // 0..511 float4 chunks
      int kr = q >> 4;                        // tile row 0..31
      int cg = (q & 15) << 2;                 // tile col 0,4,..,60
      int kg = ks*32 + kr;
      int ksrc = (kg < CIN) ? kg : (CIN - 1);
      unsigned long long ga =
          (unsigned long long)(const void*)(X + (size_t)ksrc*LSEQ + col0 + cg);
      unsigned la = ldsbase + (unsigned)((kr*64 + cg)*4);
      asm volatile("global_load_async_to_lds_b128 %0, %1, off"
                   :: "v"(la), "v"(ga) : "memory");
    }
    asm volatile("s_wait_asynccnt 0x0" ::: "memory");

    #pragma unroll
    for (int t = 0; t < NT; ++t){
      v16bf b;
      #pragma unroll
      for (int j = 0; j < 16; ++j){
        int kk = (j < 8) ? (bko + j) : (16 + bko + (j - 8));
        int kg = ks*32 + kk;
        float xv = ((CIN % 32 == 0) || (kg < CIN)) ? tw[kk*64 + t*16 + bn] : 0.0f;
        b[j] = f2bf(xv);
      }
      acc[t] = __builtin_amdgcn_wmma_f32_16x16x32_bf16(false, afrag[ks], false, b,
                                                       (short)0, acc[t], false, false);
    }
  }

  #pragma unroll
  for (int t = 0; t < NT; ++t){
    const int col = col0 + t*16 + bn;
    #pragma unroll
    for (int r = 0; r < 8; ++r){
      int m = mbase + r;
      if (m < COUT){
        float v = acc[t][r] + bias[m];
        if (ACT == 1)      v = fmaxf(v, 0.0f);
        else if (ACT == 2) v = 1.0f/(1.0f + expf(-v));
        Y[(size_t)m*LSEQ + col] = v;
      }
    }
  }
}

// ---------------------------------------------------------------- batchnorm (population stats over L)
__global__ void bn_stats_kernel(const float* __restrict__ X, float* sums, float* sumsq){
  __shared__ float s1[256], s2[256];
  int c     = blockIdx.x >> 6;
  int chunk = blockIdx.x & 63;
  const float* xr = X + (size_t)c*LSEQ + chunk*4096;
  float a = 0.0f, b = 0.0f;
  for (int k = threadIdx.x; k < 4096; k += 256){ float v = xr[k]; a += v; b += v*v; }
  s1[threadIdx.x] = a; s2[threadIdx.x] = b;
  __syncthreads();
  for (int st = 128; st > 0; st >>= 1){
    if (threadIdx.x < st){ s1[threadIdx.x] += s1[threadIdx.x+st]; s2[threadIdx.x] += s2[threadIdx.x+st]; }
    __syncthreads();
  }
  if (threadIdx.x == 0){ atomicAdd(&sums[c], s1[0]); atomicAdd(&sumsq[c], s2[0]); }
}

__global__ void bn_apply_kernel(float* X, const float* g, const float* b,
                                const float* sums, const float* sumsq, int n){
  const float invL = 1.0f/(float)LSEQ;
  for (int idx = blockIdx.x*blockDim.x + threadIdx.x; idx < n; idx += gridDim.x*blockDim.x){
    int c = idx >> 18;
    float m = sums[c]*invL;
    float v = sumsq[c]*invL - m*m;
    X[idx] = g[c]*(X[idx] - m)*rsqrtf(v + 1e-5f) + b[c];
  }
}

// ---------------------------------------------------------------- global context block
__global__ void gcb_init_kernel(float* ctx, float* scal){
  int t = threadIdx.x;
  if (t < 64) ctx[t] = 0.0f;
  if (t == 64) scal[0] = -__builtin_inff();
  if (t == 65) scal[1] = 0.0f;
}

__global__ void gcb_logits_kernel(const float* __restrict__ X, const float* __restrict__ wk,
                                  float* __restrict__ s, int C, float* maxptr){
  __shared__ float red[256];
  float lm = -__builtin_inff();
  for (int l = blockIdx.x*blockDim.x + threadIdx.x; l < LSEQ; l += gridDim.x*blockDim.x){
    float acc = 0.0f;
    for (int c = 0; c < C; ++c) acc += wk[c]*X[(size_t)c*LSEQ + l];
    s[l] = acc;
    lm = fmaxf(lm, acc);
  }
  red[threadIdx.x] = lm;
  __syncthreads();
  for (int st = 128; st > 0; st >>= 1){
    if (threadIdx.x < st) red[threadIdx.x] = fmaxf(red[threadIdx.x], red[threadIdx.x+st]);
    __syncthreads();
  }
  if (threadIdx.x == 0) atomicMaxF(maxptr, red[0]);
}

__global__ void gcb_accum_kernel(const float* __restrict__ X, const float* __restrict__ s,
                                 int C, const float* maxptr, float* ctx, float* sumE){
  __shared__ float acc[65];
  for (int i = threadIdx.x; i < 65; i += blockDim.x) acc[i] = 0.0f;
  __syncthreads();
  float mv = maxptr[0];
  for (int l = blockIdx.x*blockDim.x + threadIdx.x; l < LSEQ; l += gridDim.x*blockDim.x){
    float e = expf(s[l] - mv);
    atomicAdd(&acc[64], e);
    for (int c = 0; c < C; ++c) atomicAdd(&acc[c], X[(size_t)c*LSEQ + l]*e);
  }
  __syncthreads();
  if (threadIdx.x < C)   atomicAdd(&ctx[threadIdx.x], acc[threadIdx.x]);
  if (threadIdx.x == 64) atomicAdd(sumE, acc[64]);
}

__global__ void gcb_mlp_kernel(const float* ctx, const float* sumE,
                               const float* w1, const float* b1,
                               const float* lng, const float* lnb,
                               const float* w2, const float* b2,
                               int C, int CR, float* delta){
  __shared__ float t[8];
  __shared__ float cn[64];
  __shared__ float mv2[2];
  int tid = threadIdx.x;
  if (tid < C) cn[tid] = ctx[tid]/sumE[0];
  __syncthreads();
  if (tid < CR){
    float a = b1[tid];
    for (int c = 0; c < C; ++c) a += w1[tid*C + c]*cn[c];
    t[tid] = a;
  }
  __syncthreads();
  if (tid == 0){
    float m = 0.0f; for (int j = 0; j < CR; ++j) m += t[j]; m /= (float)CR;
    float v = 0.0f; for (int j = 0; j < CR; ++j){ float d = t[j]-m; v += d*d; } v /= (float)CR;
    mv2[0] = m; mv2[1] = rsqrtf(v + 1e-5f);
  }
  __syncthreads();
  if (tid < CR) t[tid] = fmaxf(lng[tid]*(t[tid]-mv2[0])*mv2[1] + lnb[tid], 0.0f);
  __syncthreads();
  if (tid < C){
    float a = b2[tid];
    for (int j = 0; j < CR; ++j) a += w2[tid*CR + j]*t[j];
    delta[tid] = a;
  }
}

__global__ void gcb_add_kernel(float* X, const float* delta, int n){
  for (int idx = blockIdx.x*blockDim.x + threadIdx.x; idx < n; idx += gridDim.x*blockDim.x)
    X[idx] += delta[idx >> 18];
}

// ---------------------------------------------------------------- anchor targets + loss
__global__ void gtmax_kernel(const float* __restrict__ gt, float* gtmax){
  __shared__ float sm[5];
  if (threadIdx.x < 5) sm[threadIdx.x] = 0.0f;
  __syncthreads();
  for (int i = blockIdx.x*blockDim.x + threadIdx.x; i < NANCH; i += gridDim.x*blockDim.x){
    int l = i/5, a = i - 5*l;
    float W  = (float)(8 << a);
    float as = (float)l - (W - 1.0f)*0.5f;
    float ae = (float)l + (W - 1.0f)*0.5f;
    float aw = ae - as + 1.0f;
    for (int g = 0; g < 5; ++g){
      float gs = gt[2*g], ge = gt[2*g+1], gw = ge - gs + 1.0f;
      atomicMaxF(&sm[g], iou1d(as, ae, aw, gs, ge, gw));
    }
  }
  __syncthreads();
  if (threadIdx.x < 5) atomicMaxF(&gtmax[threadIdx.x], sm[threadIdx.x]);
}

__global__ void label_loss_kernel(const float* __restrict__ probs, const float* __restrict__ deltas,
                                  const float* __restrict__ gt, const float* __restrict__ gtmax,
                                  float* accF, int* accI){
  __shared__ float rb[256], rl[256];
  __shared__ int   rm[256];
  float bce = 0.0f, l1 = 0.0f; int mc = 0;
  for (int i = blockIdx.x*blockDim.x + threadIdx.x; i < NANCH; i += gridDim.x*blockDim.x){
    int l = i/5, a = i - 5*l;
    float W  = (float)(8 << a);
    float as = (float)l - (W - 1.0f)*0.5f;
    float ae = (float)l + (W - 1.0f)*0.5f;
    float aw = ae - as + 1.0f;
    float ac = as + 0.5f*aw;
    float maxi = -1.0f; int arg = 0; int best = 0;
    for (int g = 0; g < 5; ++g){
      float gs = gt[2*g], ge = gt[2*g+1], gw = ge - gs + 1.0f;
      float io = iou1d(as, ae, aw, gs, ge, gw);
      if (io > maxi){ maxi = io; arg = g; }
      if (io >= gtmax[g]) best = 1;
    }
    int label;
    if (best || maxi >= 0.7f) label = 1;
    else if (maxi < 0.3f)     label = 0;
    else                      label = -1;
    if (label != -1){
      ++mc;
      float p = probs[(size_t)a*LSEQ + l];
      p = fminf(fmaxf(p, 1e-7f), 1.0f - 1e-7f);
      float y = (label == 1) ? 1.0f : 0.0f;
      bce += -(y*logf(p) + (1.0f - y)*log1pf(-p));
    }
    if (label == 1){
      float gs = gt[2*arg], ge = gt[2*arg+1];
      float tgw = ge - gs + 1.0f;
      float tgc = gs + 0.5f*tgw;
      float d0 = deltas[(size_t)(2*a)  *LSEQ + l] - (tgc - ac)/aw;
      float d1 = deltas[(size_t)(2*a+1)*LSEQ + l] - logf(tgw/aw);
      float a0 = fabsf(d0), a1 = fabsf(d1);
      l1 += (a0 < 1.0f) ? 0.5f*d0*d0 : (a0 - 0.5f);
      l1 += (a1 < 1.0f) ? 0.5f*d1*d1 : (a1 - 0.5f);
    }
  }
  rb[threadIdx.x] = bce; rl[threadIdx.x] = l1; rm[threadIdx.x] = mc;
  __syncthreads();
  for (int st = 128; st > 0; st >>= 1){
    if (threadIdx.x < st){
      rb[threadIdx.x] += rb[threadIdx.x+st];
      rl[threadIdx.x] += rl[threadIdx.x+st];
      rm[threadIdx.x] += rm[threadIdx.x+st];
    }
    __syncthreads();
  }
  if (threadIdx.x == 0){
    atomicAdd(&accF[0], rb[0]);
    atomicAdd(&accF[1], rl[0]);
    atomicAdd(accI, rm[0]);
  }
}

__global__ void loss_final_kernel(const float* accF, const int* accI, float* out){
  float m = (float)(accI[0] > 1 ? accI[0] : 1);
  out[0] = accF[0]/m + accF[1]/(m*(float)NANCH);
}

// ---------------------------------------------------------------- proposal layer
__global__ void boxscore_kernel(const float* __restrict__ probs, const float* __restrict__ deltas,
                                float* __restrict__ boxes, float* __restrict__ scores, int* hist){
  __shared__ int hh[256];
  for (int i = threadIdx.x; i < 256; i += blockDim.x) hh[i] = 0;
  __syncthreads();
  for (int i = blockIdx.x*blockDim.x + threadIdx.x; i < NANCH; i += gridDim.x*blockDim.x){
    int l = i/5, a = i - 5*l;
    float W  = (float)(8 << a);
    float as = (float)l - (W - 1.0f)*0.5f;
    float ae = (float)l + (W - 1.0f)*0.5f;
    float aw = ae - as + 1.0f;
    float ac = as + 0.5f*aw;
    float d0 = deltas[(size_t)(2*a)  *LSEQ + l];
    float d1 = deltas[(size_t)(2*a+1)*LSEQ + l];
    float pc = d0*aw + ac;
    float pw = expf(d1)*aw;
    float b0 = fminf(fmaxf(pc - 0.5f*pw, 0.0f), (float)(LSEQ - 1));
    float b1 = fminf(fmaxf(pc + 0.5f*pw, 0.0f), (float)(LSEQ - 1));
    float sc = probs[(size_t)a*LSEQ + l];
    boxes[2*i] = b0; boxes[2*i+1] = b1; scores[i] = sc;
    atomicAdd(&hh[__float_as_uint(sc) >> 24], 1);
  }
  __syncthreads();
  for (int i = threadIdx.x; i < 256; i += blockDim.x) if (hh[i]) atomicAdd(&hist[i], hh[i]);
}

__global__ void thresh_kernel(const int* hist, int* cnt){
  int cum = 0, boundary = 0, above = 0;
  for (int b = 255; b >= 0; --b){
    if (cum + hist[b] >= PRE_N){ boundary = b; above = cum; break; }
    cum += hist[b];
  }
  cnt[0] = boundary; cnt[1] = above; cnt[2] = PRE_N - above;
}

__global__ void compact_kernel(const float* __restrict__ scores, const float* __restrict__ boxes,
                               int* cnt, float* topS, float* topB){
  int boundary = cnt[0], above = cnt[1], quota = cnt[2];
  for (int i = blockIdx.x*blockDim.x + threadIdx.x; i < NANCH; i += gridDim.x*blockDim.x){
    float sc = scores[i];
    int bin = (int)(__float_as_uint(sc) >> 24);
    int slot = -1;
    if (bin > boundary) slot = atomicAdd(&cnt[3], 1);
    else if (bin == boundary){
      int q = atomicAdd(&cnt[4], 1);
      if (q < quota) slot = above + q;
    }
    if (slot >= 0 && slot < PRE_N){
      topS[slot] = sc;
      topB[2*slot]   = boxes[2*i];
      topB[2*slot+1] = boxes[2*i+1];
    }
  }
}

__global__ void sort_gather_kernel(const float* topS, const float* topB,
                                   float* srtS, float* srtB){
  __shared__ float ss[SORT_N];
  __shared__ short si[SORT_N];
  for (int i = threadIdx.x; i < SORT_N; i += blockDim.x){
    if (i < PRE_N){ ss[i] = topS[i]; si[i] = (short)i; }
    else          { ss[i] = -1.0f;   si[i] = -1; }
  }
  for (int k = 2; k <= SORT_N; k <<= 1){
    for (int j = k >> 1; j > 0; j >>= 1){
      __syncthreads();
      for (int i = threadIdx.x; i < SORT_N; i += blockDim.x){
        int ixj = i ^ j;
        if (ixj > i){
          bool up = ((i & k) == 0);               // descending overall
          float a = ss[i], b = ss[ixj];
          bool sw = up ? (a < b) : (a > b);
          if (sw){
            ss[i] = b; ss[ixj] = a;
            short t = si[i]; si[i] = si[ixj]; si[ixj] = t;
          }
        }
      }
    }
  }
  __syncthreads();
  for (int i = threadIdx.x; i < PRE_N; i += blockDim.x){
    srtS[i] = ss[i];
    int id = si[i];
    srtB[2*i]   = (id >= 0) ? topB[2*id]   : 0.0f;
    srtB[2*i+1] = (id >= 0) ? topB[2*id+1] : 0.0f;
  }
}

__global__ void nms_kernel(const float* __restrict__ srtB, int* keepOut){
  __shared__ float bs[PRE_N], be[PRE_N];
  __shared__ unsigned char kp[PRE_N];
  for (int i = threadIdx.x; i < PRE_N; i += blockDim.x){
    bs[i] = srtB[2*i]; be[i] = srtB[2*i+1]; kp[i] = 1;
  }
  __syncthreads();
  for (int i = 0; i < PRE_N; ++i){
    if (kp[i]){
      float si_ = bs[i], ei = be[i];
      float wi = ei - si_ + 1.0f;
      for (int j = i + 1 + (int)threadIdx.x; j < PRE_N; j += blockDim.x){
        float inter = fminf(ei, be[j]) - fmaxf(si_, bs[j]) + 1.0f;
        inter = fmaxf(inter, 0.0f);
        float wj = be[j] - bs[j] + 1.0f;
        if (inter/(wi + wj - inter) > 0.7f) kp[j] = 0;
      }
    }
    __syncthreads();
  }
  for (int i = threadIdx.x; i < PRE_N; i += blockDim.x) keepOut[i] = (int)kp[i];
}

__global__ void output_kernel(const float* srtS, const float* srtB, const int* keep, float* out){
  int r = 0;
  for (int i = 0; i < PRE_N && r < POST_N; ++i)
    if (keep[i]){ out[1+2*r] = srtB[2*i]; out[2+2*r] = srtB[2*i+1]; out[601+r] = srtS[i]; ++r; }
  for (int i = 0; i < PRE_N && r < POST_N; ++i)
    if (!keep[i]){ out[1+2*r] = srtB[2*i]; out[2+2*r] = srtB[2*i+1]; out[601+r] = srtS[i]; ++r; }
}

// ================================================================ host side
struct BlockP { const float *b1,*b2,*bnb,*bng,*dw,*lnb,*lng,*pw,*pwb,*w1,*w2,*wk; };

static BlockP grab_block(void* const* d_in, int& i){
  BlockP p;
  p.b1  = (const float*)d_in[i++]; p.b2  = (const float*)d_in[i++];
  p.bnb = (const float*)d_in[i++]; p.bng = (const float*)d_in[i++];
  p.dw  = (const float*)d_in[i++]; p.lnb = (const float*)d_in[i++];
  p.lng = (const float*)d_in[i++]; p.pw  = (const float*)d_in[i++];
  p.pwb = (const float*)d_in[i++]; p.w1  = (const float*)d_in[i++];
  p.w2  = (const float*)d_in[i++]; p.wk  = (const float*)d_in[i++];
  return p;
}

typedef void (*PwFn)(const float*, const float*, const float*, float*, hipStream_t);

template<int CIN, int COUT, int ACT>
static void launch_pw_t(const float* X, const float* W, const float* b, float* Y, hipStream_t s){
  constexpr int MTILES = (COUT + 15)/16;
  int waves = MTILES*(LSEQ/64);
  pwconv_wmma_t<CIN, COUT, ACT><<<(waves + 7)/8, 256, 0, s>>>(X, W, b, Y);
}

static void run_block(const float* in, const BlockP& p, int cin, int cout, int pad, int dil, int cr,
                      PwFn pw, float* X, float* Yb, float* SB,
                      float* sums, float* ctx, float* scal, float* delta, hipStream_t s){
  int nin = cin*LSEQ;
  dwconv_kernel<<<(nin + 255)/256, 256, 0, s>>>(in, p.dw, Yb, pad, dil, nin);
  pw(Yb, p.pw, p.pwb, X, s);
  fill_kernel<<<1, 128, 0, s>>>(sums, 0.0f, 128);           // sums + sumsq
  bn_stats_kernel<<<cout*64, 256, 0, s>>>(X, sums, sums + 64);
  int n = cout*LSEQ;
  bn_apply_kernel<<<(n + 255)/256, 256, 0, s>>>(X, p.bng, p.bnb, sums, sums + 64, n);
  gcb_init_kernel<<<1, 128, 0, s>>>(ctx, scal);
  gcb_logits_kernel<<<256, 256, 0, s>>>(X, p.wk, SB, cout, scal);
  gcb_accum_kernel<<<256, 256, 0, s>>>(X, SB, cout, scal, ctx, scal + 1);
  gcb_mlp_kernel<<<1, 64, 0, s>>>(ctx, scal + 1, p.w1, p.b1, p.lng, p.lnb, p.w2, p.b2, cout, cr, delta);
  gcb_add_kernel<<<(n + 255)/256, 256, 0, s>>>(X, delta, n);
}

extern "C" void kernel_launch(void* const* d_in, const int* in_sizes, int n_in,
                              void* d_out, int out_size, void* d_ws, size_t ws_size,
                              hipStream_t stream){
  (void)in_sizes; (void)n_in; (void)out_size; (void)ws_size;
  const size_t L = LSEQ;
  float* ws  = (float*)d_ws;
  float* out = (float*)d_out;

  // ---- inputs (top dict order, params tree-flattened with sorted dict keys)
  int ii = 0;
  const float* seq    = (const float*)d_in[ii++];
  const float* gt     = (const float*)d_in[ii++];
  const float* bbox_b = (const float*)d_in[ii++];
  const float* bbox_w = (const float*)d_in[ii++];
  const float* cls_b  = (const float*)d_in[ii++];
  const float* cls_w  = (const float*)d_in[ii++];
  BlockP dec[5], enc[5];
  for (int i = 0; i < 5; ++i) dec[i] = grab_block(d_in, ii);
  for (int i = 0; i < 5; ++i) enc[i] = grab_block(d_in, ii);
  const float* rpn_bnb = (const float*)d_in[ii++];
  const float* rpn_bng = (const float*)d_in[ii++];
  const float* rpn_dw  = (const float*)d_in[ii++];
  const float* rpn_pw  = (const float*)d_in[ii++];
  const float* rpn_pwb = (const float*)d_in[ii++];

  // ---- workspace layout (~199 MB)
  float* X  = ws;                 // 64L  current feature map (channel-major)
  float* Yb = ws + 64*L;          // 64L  dwconv scratch / probs+deltas later
  float* SK[4] = { ws + 128*L, ws + 160*L, ws + 176*L, ws + 184*L };  // skips 32/16/8/4 ch
  float* SB   = ws + 188*L;       // L    gcb logits
  float* TOPS = ws + 189*L;                 // 8192
  float* TOPB = TOPS + SORT_N;              // 16384
  float* SRTS = TOPB + 2*SORT_N;            // 8192
  float* SRTB = SRTS + SORT_N;              // 16384
  int*   KEEP = (int*)(SRTB + 2*SORT_N);    // 8192 ints
  float* SM   = (float*)(KEEP + SORT_N);
  float* sums = SM;               // 64 (+64 sumsq)
  float* ctx  = SM + 128;         // 64
  float* delta= SM + 192;         // 64
  float* scal = SM + 256;         // [0]=softmax max, [1]=sumE
  float* gmax = SM + 264;         // 5 (pad 8)
  float* accF = SM + 272;         // bce, l1
  int*   accI = (int*)(SM + 274); // mask count
  int*   hist = (int*)(SM + 288); // 256
  int*   cnt  = (int*)(SM + 544); // boundary, above, quota, cnt_hi, cnt_eq
  float* BOXES  = ws + 128*L;     // 10L (overlays dead skips)
  float* SCORES = ws + 138*L;     // 5L
  float* PROBS  = Yb;             // 5L  (overlays dead dwconv scratch)
  float* DELTAS = Yb + 5*L;       // 10L

  // ---- encoder
  static const int ECI[5] = {14,32,16, 8, 4}, ECO[5] = {32,16, 8, 4, 2};
  static const int EP [5] = { 1, 1, 2, 2, 3}, ED [5] = { 1, 1, 2, 2, 3};
  static const PwFn EPW[5] = { launch_pw_t<14,32,1>, launch_pw_t<32,16,1>,
                               launch_pw_t<16, 8,1>, launch_pw_t< 8, 4,1>,
                               launch_pw_t< 4, 2,1> };
  const float* cur = seq;
  for (int i = 0; i < 5; ++i){
    run_block(cur, enc[i], ECI[i], ECO[i], EP[i], ED[i], 2, EPW[i],
              X, Yb, SB, sums, ctx, scal, delta, stream);
    cur = X;
    if (i < 4) hipMemcpyAsync(SK[i], X, (size_t)ECO[i]*L*sizeof(float), hipMemcpyDeviceToDevice, stream);
  }

  // ---- decoder (block output in X[0..cout), skip appended at channel offset cout)
  static const int DCI[5] = { 2, 8,16,32,64}, DCO[5] = { 4, 8,16,32,32};
  static const int DP [5] = { 3, 2, 2, 1, 1}, DD [5] = { 3, 2, 2, 1, 1};
  static const int DCR[5] = { 4, 4, 4, 4, 2};
  static const PwFn DPW[5] = { launch_pw_t< 2, 4,1>, launch_pw_t< 8, 8,1>,
                               launch_pw_t<16,16,1>, launch_pw_t<32,32,1>,
                               launch_pw_t<64,32,1> };
  for (int i = 0; i < 5; ++i){
    run_block(X, dec[i], DCI[i], DCO[i], DP[i], DD[i], DCR[i], DPW[i],
              X, Yb, SB, sums, ctx, scal, delta, stream);
    if (i < 4){
      int sidx = 3 - i;  // skips popped LIFO: e3, e2, e1, e0
      hipMemcpyAsync(X + (size_t)DCO[i]*L, SK[sidx], (size_t)ECO[sidx]*L*sizeof(float),
                     hipMemcpyDeviceToDevice, stream);
    }
  }

  // ---- RPN head: fm (32ch) in X
  int n32 = 32*LSEQ;
  dwconv_kernel<<<(n32 + 255)/256, 256, 0, stream>>>(X, rpn_dw, Yb, 1, 1, n32);
  launch_pw_t<32,16,1>(Yb, rpn_pw, rpn_pwb, X, stream);
  fill_kernel<<<1, 128, 0, stream>>>(sums, 0.0f, 128);
  bn_stats_kernel<<<16*64, 256, 0, stream>>>(X, sums, sums + 64);
  int n16 = 16*LSEQ;
  bn_apply_kernel<<<(n16 + 255)/256, 256, 0, stream>>>(X, rpn_bng, rpn_bnb, sums, sums + 64, n16);
  launch_pw_t<16, 5,2>(X, cls_w,  cls_b,  PROBS,  stream);   // sigmoid
  launch_pw_t<16,10,0>(X, bbox_w, bbox_b, DELTAS, stream);   // linear

  // ---- anchor targets + loss -> out[0]
  fill_kernel<<<1, 64, 0, stream>>>(gmax, 0.0f, 8);
  fill_kernel<<<1, 64, 0, stream>>>(accF, 0.0f, 4);
  gtmax_kernel<<<2048, 256, 0, stream>>>(gt, gmax);
  label_loss_kernel<<<2048, 256, 0, stream>>>(PROBS, DELTAS, gt, gmax, accF, accI);
  loss_final_kernel<<<1, 1, 0, stream>>>(accF, accI, out);

  // ---- proposals: histogram top-k -> sort -> NMS -> 300
  fill_kernel<<<1, 256, 0, stream>>>((float*)hist, 0.0f, 256);
  fill_kernel<<<1, 64, 0, stream>>>((float*)cnt, 0.0f, 8);
  boxscore_kernel<<<2048, 256, 0, stream>>>(PROBS, DELTAS, BOXES, SCORES, hist);
  thresh_kernel<<<1, 1, 0, stream>>>(hist, cnt);
  compact_kernel<<<2048, 256, 0, stream>>>(SCORES, BOXES, cnt, TOPS, TOPB);
  sort_gather_kernel<<<1, 1024, 0, stream>>>(TOPS, TOPB, SRTS, SRTB);
  nms_kernel<<<1, 1024, 0, stream>>>(SRTB, KEEP);
  output_kernel<<<1, 1, 0, stream>>>(SRTS, SRTB, KEEP, out);
}
